// DistributionalQNetwork_11373073400424
// MI455X (gfx1250) — matrix operand order repaired and checked by reading, not compiled, not run on hardware
//
#include <hip/hip_runtime.h>
#include <cstdint>

typedef __attribute__((ext_vector_type(16))) __bf16 bf16x16;
typedef __attribute__((ext_vector_type(8)))  __bf16 bf16x8;
typedef __attribute__((ext_vector_type(8)))  float  f32x8;

#define NATOMS 101
#define BATCH  65536

// ---------------------------------------------------------------------------
// CDNA5 async global->LDS copy (ASYNCcnt-tracked, bypasses VGPRs).
// GVS saddr form: SGPR-pair base + 32-bit VGPR byte offset -> 1 address VGPR.
// LDS operand: low 32 bits of a generic pointer to a __shared__ object are
// the wave-relative LDS byte address the instruction wants.
// ---------------------------------------------------------------------------
__device__ __forceinline__ void async_b128(const void* sbase, uint32_t voff_bytes,
                                           void* l) {
  asm volatile("global_load_async_to_lds_b128 %0, %1, %2"
               :
               : "v"((uint32_t)(uintptr_t)l), "v"(voff_bytes),
                 "s"((uint64_t)(uintptr_t)sbase)
               : "memory");
}
__device__ __forceinline__ void wait_async0() {
  asm volatile("s_wait_asynccnt 0" ::: "memory");
}

// ---------------------------------------------------------------------------
// Weight prep: W[K][N] (f32) -> Wt[Nout][Kpad] (bf16, transposed, zero-pad).
// ---------------------------------------------------------------------------
__global__ __launch_bounds__(256) void wprep_kernel(
    const float* __restrict__ W, __bf16* __restrict__ Wt,
    int K, int N, int Kpad, int Nout)
{
  int idx = blockIdx.x * 256 + threadIdx.x;
  int total = Nout * Kpad;
  if (idx >= total) return;
  int n = idx / Kpad;
  int k = idx - n * Kpad;
  float v = 0.0f;
  if (k < K && n < N) v = W[(size_t)k * N + n];
  Wt[idx] = (__bf16)v;
}

// ---------------------------------------------------------------------------
// Layer-1 GEMM, double-buffered, 32-bit offset addressing.
//  A tile: fused concat(obs,act) f32 -> bf16, staged in registers across the
//          WMMA block so the global-load wait lands after the matrix ops.
//  B tile: async global->LDS.
// ---------------------------------------------------------------------------
template<int NT>
__global__ __launch_bounds__(256) void gemm_first_kernel(
    const float* __restrict__ obs, const float* __restrict__ act,
    const __bf16* __restrict__ Wt, const float* __restrict__ bias,
    __bf16* __restrict__ O, int KT, int Kpad, int Nout)
{
  constexpr int NBLK = NT * 16;
  __shared__ alignas(16) __bf16 As[2][128][40];   // rows padded: bank-conflict free
  __shared__ alignas(16) __bf16 Bs[2][NBLK][40];

  const int tid  = threadIdx.x;
  const int lane = tid & 31;
  const int wave = tid >> 5;
  const int half = lane >> 4;
  const int l16  = lane & 15;
  const uint32_t m0 = blockIdx.y * 128;
  const uint32_t n0 = blockIdx.x * NBLK;

  const f32x8 fzero = {0.f,0.f,0.f,0.f,0.f,0.f,0.f,0.f};
  f32x8 acc[NT];
#pragma unroll
  for (int nt = 0; nt < NT; ++nt) acc[nt] = fzero;

  auto loadA_regs = [&](int kt, float* areg) {
    const uint32_t k0 = (uint32_t)kt * 32u;
    if (k0 + 32u <= 1090u) {               // fast path: pure obs columns
#pragma unroll
      for (int i = 0; i < 16; ++i) {
        uint32_t idx = (uint32_t)tid + (uint32_t)i * 256u;
        uint32_t r = idx >> 5, c = idx & 31u;
        areg[i] = obs[(m0 + r) * 1090u + (k0 + c)];
      }
    } else {                               // boundary: obs | act | zero-pad
#pragma unroll
      for (int i = 0; i < 16; ++i) {
        uint32_t idx = (uint32_t)tid + (uint32_t)i * 256u;
        uint32_t r = idx >> 5, c = idx & 31u;
        uint32_t gk = k0 + c, m = m0 + r;
        float v = 0.0f;
        if (gk < 1090u)      v = obs[m * 1090u + gk];
        else if (gk < 1092u) v = act[m * 2u + (gk - 1090u)];
        areg[i] = v;
      }
    }
  };
  auto storeA_regs = [&](const float* areg, int buf) {
#pragma unroll
    for (int i = 0; i < 16; ++i) {
      uint32_t idx = (uint32_t)tid + (uint32_t)i * 256u;
      uint32_t r = idx >> 5, c = idx & 31u;
      As[buf][r][c] = (__bf16)areg[i];
    }
  };
  auto fillB_async = [&](int kt, int buf) {
    const uint32_t k0 = (uint32_t)kt * 32u;
    for (int ch = tid; ch < NBLK * 4; ch += 256) {
      uint32_t rr = (uint32_t)ch >> 2, c8 = ((uint32_t)ch & 3u) * 8u;
      uint32_t off = ((n0 + rr) * (uint32_t)Kpad + k0 + c8) * 2u;   // bytes
      async_b128(Wt, off, &Bs[buf][rr][c8]);
    }
  };

  { // prologue: stage tile 0 into buffer 0
    float areg[16];
    loadA_regs(0, areg);
    storeA_regs(areg, 0);
    fillB_async(0, 0);
    wait_async0();
  }
  __syncthreads();

  for (int kt = 0; kt < KT; ++kt) {
    const int p = kt & 1;
    const bool nxt = (kt + 1 < KT);
    if (nxt) fillB_async(kt + 1, 1 - p);
    float areg[16];
    if (nxt) loadA_regs(kt + 1, areg);

    // ---- WMMA block on buffer p ----
    bf16x16 afrag;
    {
      const int arow = wave * 16 + l16;
      bf16x8 alo = *(const bf16x8*)&As[p][arow][half * 16];
      bf16x8 ahi = *(const bf16x8*)&As[p][arow][half * 16 + 8];
#pragma unroll
      for (int i = 0; i < 8; ++i) { afrag[i] = alo[i]; afrag[8 + i] = ahi[i]; }
    }
    // B fragments software-pipelined one tile ahead so DS latency overlaps WMMA
    bf16x8 blo = *(const bf16x8*)&Bs[p][l16][half * 16];
    bf16x8 bhi = *(const bf16x8*)&Bs[p][l16][half * 16 + 8];
#pragma unroll
    for (int nt = 0; nt < NT; ++nt) {
      bf16x8 nlo = blo, nhi = bhi;
      if (nt + 1 < NT) {
        nlo = *(const bf16x8*)&Bs[p][(nt + 1) * 16 + l16][half * 16];
        nhi = *(const bf16x8*)&Bs[p][(nt + 1) * 16 + l16][half * 16 + 8];
      }
      bf16x16 bfrag;
#pragma unroll
      for (int i = 0; i < 8; ++i) { bfrag[i] = blo[i]; bfrag[8 + i] = bhi[i]; }
      acc[nt] = __builtin_amdgcn_wmma_f32_16x16x32_bf16(
          false, afrag, false, bfrag, (short)0, acc[nt], false, false);
      blo = nlo; bhi = nhi;
    }
    // ---- stage next A tile after the WMMAs ----
    if (nxt) storeA_regs(areg, 1 - p);
    wait_async0();
    __syncthreads();
  }

  // epilogue: +bias, relu, -> bf16. D layout: row = half*8 + r, col = l16.
#pragma unroll
  for (int nt = 0; nt < NT; ++nt) {
    uint32_t n = n0 + (uint32_t)(nt * 16 + l16);
    float bv = bias[n];
#pragma unroll
    for (int r = 0; r < 8; ++r) {
      uint32_t m = m0 + (uint32_t)(wave * 16 + half * 8 + r);
      float v = fmaxf(acc[nt][r] + bv, 0.0f);
      O[m * (uint32_t)Nout + n] = (__bf16)v;
    }
  }
}

// ---------------------------------------------------------------------------
// Mid/last layer GEMM: A already bf16 -> both tiles stream via async
// global->LDS (saddr form), 2-deep ping-pong, 32-bit offsets.
// ---------------------------------------------------------------------------
template<int NT, bool RELU, bool OUT_F32>
__global__ __launch_bounds__(256) void gemm_bf16_kernel(
    const __bf16* __restrict__ A, int K,
    const __bf16* __restrict__ Wt, int Kpad,
    const float* __restrict__ bias, int NBiasValid,
    __bf16* __restrict__ Obf, float* __restrict__ Of32,
    int Nout, int KT)
{
  constexpr int NBLK = NT * 16;
  __shared__ alignas(16) __bf16 As[2][128][40];
  __shared__ alignas(16) __bf16 Bs[2][NBLK][40];

  const int tid  = threadIdx.x;
  const int lane = tid & 31;
  const int wave = tid >> 5;
  const int half = lane >> 4;
  const int l16  = lane & 15;
  const uint32_t m0 = blockIdx.y * 128;
  const uint32_t n0 = blockIdx.x * NBLK;

  const f32x8 fzero = {0.f,0.f,0.f,0.f,0.f,0.f,0.f,0.f};
  f32x8 acc[NT];
#pragma unroll
  for (int nt = 0; nt < NT; ++nt) acc[nt] = fzero;

  auto fill_async = [&](int kt, int buf) {
    const uint32_t k0 = (uint32_t)kt * 32u;
    for (int ch = tid; ch < 128 * 4; ch += 256) {        // A tile, 16B chunks
      uint32_t r = (uint32_t)ch >> 2, c8 = ((uint32_t)ch & 3u) * 8u;
      uint32_t off = ((m0 + r) * (uint32_t)K + k0 + c8) * 2u;
      async_b128(A, off, &As[buf][r][c8]);
    }
    for (int ch = tid; ch < NBLK * 4; ch += 256) {       // B tile
      uint32_t rr = (uint32_t)ch >> 2, c8 = ((uint32_t)ch & 3u) * 8u;
      uint32_t off = ((n0 + rr) * (uint32_t)Kpad + k0 + c8) * 2u;
      async_b128(Wt, off, &Bs[buf][rr][c8]);
    }
  };

  fill_async(0, 0);
  wait_async0();
  __syncthreads();

  for (int kt = 0; kt < KT; ++kt) {
    const int p = kt & 1;
    if (kt + 1 < KT) fill_async(kt + 1, 1 - p);

    bf16x16 afrag;
    {
      const int arow = wave * 16 + l16;
      bf16x8 alo = *(const bf16x8*)&As[p][arow][half * 16];
      bf16x8 ahi = *(const bf16x8*)&As[p][arow][half * 16 + 8];
#pragma unroll
      for (int i = 0; i < 8; ++i) { afrag[i] = alo[i]; afrag[8 + i] = ahi[i]; }
    }
    bf16x8 blo = *(const bf16x8*)&Bs[p][l16][half * 16];
    bf16x8 bhi = *(const bf16x8*)&Bs[p][l16][half * 16 + 8];
#pragma unroll
    for (int nt = 0; nt < NT; ++nt) {
      bf16x8 nlo = blo, nhi = bhi;
      if (nt + 1 < NT) {
        nlo = *(const bf16x8*)&Bs[p][(nt + 1) * 16 + l16][half * 16];
        nhi = *(const bf16x8*)&Bs[p][(nt + 1) * 16 + l16][half * 16 + 8];
      }
      bf16x16 bfrag;
#pragma unroll
      for (int i = 0; i < 8; ++i) { bfrag[i] = blo[i]; bfrag[8 + i] = bhi[i]; }
      acc[nt] = __builtin_amdgcn_wmma_f32_16x16x32_bf16(
          false, afrag, false, bfrag, (short)0, acc[nt], false, false);
      blo = nlo; bhi = nhi;
    }
    wait_async0();
    __syncthreads();
  }

#pragma unroll
  for (int nt = 0; nt < NT; ++nt) {
    uint32_t n = n0 + (uint32_t)(nt * 16 + l16);
    float bv = ((int)n < NBiasValid) ? bias[n] : 0.0f;
#pragma unroll
    for (int r = 0; r < 8; ++r) {
      uint32_t m = m0 + (uint32_t)(wave * 16 + half * 8 + r);
      float v = acc[nt][r] + bv;
      if (RELU) v = fmaxf(v, 0.0f);
      if (OUT_F32) Of32[m * (uint32_t)Nout + n] = v;
      else         Obf[m * (uint32_t)Nout + n] = (__bf16)v;
    }
  }
}

// ---------------------------------------------------------------------------
// Softmax + C51 categorical projection (exact reference mask semantics).
// ---------------------------------------------------------------------------
__global__ __launch_bounds__(128) void c51_kernel(
    const float* __restrict__ LG,      // [B][112] logits (cols 101..111 junk)
    const float* __restrict__ rew, const float* __restrict__ boot,
    const float* __restrict__ disc, const float* __restrict__ qsup,
    float* __restrict__ out)           // [B][101]
{
  __shared__ float qs[NATOMS];
  __shared__ float bins[128][104];

  const int tid = threadIdx.x;
  if (tid < NATOMS) qs[tid] = qsup[tid];
  __syncthreads();

  const uint32_t m = blockIdx.x * 128 + tid;
  const float* lrow = LG + m * 112u;

  float mx = -3.402823466e38f;
  for (int j = 0; j < NATOMS; ++j) mx = fmaxf(mx, lrow[j]);
  float s = 0.0f;
  for (int j = 0; j < NATOMS; ++j) s += __expf(lrow[j] - mx);
  const float inv = 1.0f / s;

  const float r  = rew[m];
  const float bd = boot[m] * disc[m];

  for (int j = 0; j < NATOMS; ++j) bins[tid][j] = 0.0f;

  for (int j = 0; j < NATOMS; ++j) {
    float p  = __expf(lrow[j] - mx) * inv;
    float tz = fminf(fmaxf(r + bd * qs[j], -10.0f), 10.0f);
    float b  = (tz + 10.0f) / 0.2f;          // in [0, 100]
    int li0 = (int)floorf(b);
    int ui0 = (int)ceilf(b);
    bool eq = (li0 == ui0);
    int li = (eq && (ui0 > 0))          ? li0 - 1 : li0;
    int ui = (eq && (li0 < NATOMS - 1)) ? ui0 + 1 : ui0;
    bins[tid][li] += p * ((float)ui - b);
    bins[tid][ui] += p * (b - (float)li);
  }
  __syncthreads();

  const int total = 128 * NATOMS;
  const size_t base = (size_t)blockIdx.x * total;
  for (int idx = tid; idx < total; idx += 128) {
    int rr = idx / NATOMS;
    int c  = idx - rr * NATOMS;
    out[base + idx] = bins[rr][c];
  }
}

// ---------------------------------------------------------------------------
extern "C" void kernel_launch(void* const* d_in, const int* in_sizes, int n_in,
                              void* d_out, int out_size, void* d_ws, size_t ws_size,
                              hipStream_t stream) {
  (void)in_sizes; (void)n_in; (void)out_size; (void)ws_size;
  const float* obs  = (const float*)d_in[0];
  const float* act  = (const float*)d_in[1];
  const float* rew  = (const float*)d_in[2];
  const float* boot = (const float*)d_in[3];
  const float* disc = (const float*)d_in[4];
  const float* qsup = (const float*)d_in[5];
  const float* W1 = (const float*)d_in[6];   const float* b1 = (const float*)d_in[7];
  const float* W2 = (const float*)d_in[8];   const float* b2 = (const float*)d_in[9];
  const float* W3 = (const float*)d_in[10];  const float* b3 = (const float*)d_in[11];
  const float* W4 = (const float*)d_in[12];  const float* b4 = (const float*)d_in[13];
  float* out = (float*)d_out;
  char* ws = (char*)d_ws;

  size_t off = 0;
  auto take = [&](size_t bytes) {
    size_t o = off;
    off += (bytes + 255) & ~(size_t)255;
    return o;
  };
  __bf16* Wt1 = (__bf16*)(ws + take((size_t)512 * 1120 * 2));
  __bf16* Wt2 = (__bf16*)(ws + take((size_t)256 * 512  * 2));
  __bf16* Wt3 = (__bf16*)(ws + take((size_t)128 * 256  * 2));
  __bf16* Wt4 = (__bf16*)(ws + take((size_t)112 * 128  * 2));
  __bf16* H1  = (__bf16*)(ws + take((size_t)BATCH * 512 * 2));
  __bf16* H2  = (__bf16*)(ws + take((size_t)BATCH * 256 * 2));
  __bf16* H3  = (__bf16*)(ws + take((size_t)BATCH * 128 * 2));
  float*  LG  = (float*) (ws + take((size_t)BATCH * 112 * 4));

  auto blocks = [](int total) { return (total + 255) / 256; };
  wprep_kernel<<<blocks(512 * 1120), 256, 0, stream>>>(W1, Wt1, 1092, 512, 1120, 512);
  wprep_kernel<<<blocks(256 * 512),  256, 0, stream>>>(W2, Wt2, 512,  256, 512,  256);
  wprep_kernel<<<blocks(128 * 256),  256, 0, stream>>>(W3, Wt3, 256,  128, 256,  128);
  wprep_kernel<<<blocks(112 * 128),  256, 0, stream>>>(W4, Wt4, 128,  101, 128,  112);

  gemm_first_kernel<16><<<dim3(2, BATCH / 128), 256, 0, stream>>>(
      obs, act, Wt1, b1, H1, /*KT=*/35, /*Kpad=*/1120, /*Nout=*/512);

  gemm_bf16_kernel<16, true, false><<<dim3(1, BATCH / 128), 256, 0, stream>>>(
      H1, 512, Wt2, 512, b2, 256, H2, nullptr, 256, 16);

  gemm_bf16_kernel<8, true, false><<<dim3(1, BATCH / 128), 256, 0, stream>>>(
      H2, 256, Wt3, 256, b3, 128, H3, nullptr, 128, 8);

  gemm_bf16_kernel<7, false, true><<<dim3(1, BATCH / 128), 256, 0, stream>>>(
      H3, 128, Wt4, 128, b4, 101, nullptr, LG, 112, 4);

  c51_kernel<<<BATCH / 128, 128, 0, stream>>>(LG, rew, boot, disc, qsup, out);
}
